// MCGCL_81827716923994
// MI455X (gfx1250) — compile-verified, matrix operand order checked by reading, not run on hardware
//
#include <hip/hip_runtime.h>
#include <hip/hip_bf16.h>
#include <math.h>

// ---------------- problem constants (from reference) ----------------
#define Sn   8000
#define ENn  8000
#define Kn   1024
#define Dn   128
#define Bn   4096
#define ESK  160000
#define EEK  32000
#define EKK  20480
#define EEE  128000
#define ESS  128000

// async global->LDS staging in the GEMM (gfx1250 ASYNCcnt path).
#ifndef USE_ASYNC_LDS
#define USE_ASYNC_LDS 1
#endif

typedef __attribute__((ext_vector_type(16))) __bf16        v16bf;
typedef __attribute__((ext_vector_type(8)))  float         v8f;
typedef __attribute__((ext_vector_type(8)))  unsigned int  v8u;
typedef unsigned short u16;
typedef unsigned int   u32;

// ---------------- device helpers ----------------
__device__ __forceinline__ u16 f2bf(float f) {
    u32 u = __float_as_uint(f);
    u32 r = ((u >> 16) & 1u) + 0x7fffu;      // round-to-nearest-even
    return (u16)((u + r) >> 16);
}
__device__ __forceinline__ float bf2f(u16 h) {
    return __uint_as_float(((u32)h) << 16);
}
__device__ __forceinline__ float softplusf(float x) {
    if (x > 20.f)  return x;
    if (x < -20.f) return expf(x);
    return log1pf(expf(x));
}
__device__ __forceinline__ float leaky02(float x) { return x > 0.f ? x : 0.2f * x; }

__device__ __forceinline__ void atomicMaxF(float* a, float v) {
    int* ai = (int*)a;
    int old = *ai;
    while (__int_as_float(old) < v) {
        int assumed = old;
        old = atomicCAS(ai, assumed, __float_as_int(v));
        if (old == assumed) break;
    }
}

// Load one 16x32 bf16 WMMA fragment (A-layout; also used for rows of B^T).
// Lane l: r = l&15 (row), h = l>>4; VGPR v holds K = 8h+2v (v<4) else 16+8h+2(v-4), and K+1.
__device__ __forceinline__ v16bf load_frag16x32(const u16* p, int k0, int h) {
    v8u u;
#pragma unroll
    for (int v = 0; v < 8; ++v) {
        int kk = (v < 4) ? (8 * h + 2 * v) : (16 + 8 * h + 2 * (v - 4));
        u[v] = *(const u32*)(p + k0 + kk);
    }
    return __builtin_bit_cast(v16bf, u);
}

// ---------------- elementwise / fill kernels ----------------
__global__ void k_fill_f32(float* p, float v, size_t n) {
    size_t i = (size_t)blockIdx.x * blockDim.x + threadIdx.x;
    if (i < n) p[i] = v;
}
__global__ void k_fill_u32(u32* p, u32 v, size_t n) {
    size_t i = (size_t)blockIdx.x * blockDim.x + threadIdx.x;
    if (i < n) p[i] = v;
}
__global__ void k_to_bf16(u16* dst, const float* src, size_t n) {
    size_t i = (size_t)blockIdx.x * blockDim.x + threadIdx.x;
    if (i < n) dst[i] = f2bf(src[i]);
}
// dst[C x R] (bf16) = transpose of src[R x C] (f32)  -> builds B^T operands
__global__ void k_transpose_bf16(u16* dst, const float* src, int R, int C) {
    int i = blockIdx.x * blockDim.x + threadIdx.x;
    if (i >= R * C) return;
    int r = i / C, c = i % C;
    dst[(size_t)c * R + r] = f2bf(src[i]);
}
// dst[n x 256] bf16 = concat(a[n x 128], b[n x 128])
__global__ void k_concat2_bf16(u16* dst, const float* a, const float* b, size_t rows) {
    size_t i = (size_t)blockIdx.x * blockDim.x + threadIdx.x;
    if (i >= rows * 256) return;
    size_t r = i >> 8; int c = (int)(i & 255);
    float v = (c < 128) ? a[r * 128 + c] : b[r * 128 + (c - 128)];
    dst[i] = f2bf(v);
}
__global__ void k_relu(float* p, size_t n) {
    size_t i = (size_t)blockIdx.x * blockDim.x + threadIdx.x;
    if (i < n) p[i] = fmaxf(p[i], 0.f);
}
__global__ void k_sigmoid(float* p, size_t n) {
    size_t i = (size_t)blockIdx.x * blockDim.x + threadIdx.x;
    if (i < n) p[i] = 1.f / (1.f + expf(-p[i]));
}
__global__ void k_rows_bias(float* out, const float* bias, int n) {
    int r = blockIdx.x, d = threadIdx.x;                 // block = 128
    if (r < n) out[(size_t)r * 128 + d] = bias[d];
}

// ---------------- WMMA GEMM: C[M,N](f32) = A[M,K](bf16) * Bt[N,K](bf16)^T ----------------
// Block = 128 threads (4 waves); each block: 64 rows x 32 cols of C.
// Each wave computes TWO 16x16 C tiles sharing one A fragment -> two independent
// WMMA accumulator chains fill each other's data-hazard slots (5-slot bf16 RAW).
// The 32 x K Bt panel is staged once into LDS (async global->LDS, ASYNCcnt).
// act: 0=none 1=relu 2=tanh ; bias optional (per output column)
__global__ void k_gemm_bf16(float* __restrict__ C, const u16* __restrict__ A,
                            const u16* __restrict__ Bt, const float* __restrict__ bias,
                            int M, int N, int K, int act) {
    __shared__ __align__(16) u16 sB[32 * 256];           // K <= 256 -> max 16KB
    int tid  = threadIdx.x;
    int wave = tid >> 5;
    int lane = tid & 31;
    int n0 = blockIdx.x * 32;
    int m0 = (blockIdx.y * 4 + wave) * 16;

    // ---- cooperative stage: sB[0:32, 0:K] = Bt[n0:n0+32, 0:K] ----
    const u16* gB = Bt + (size_t)n0 * K;
    int chunks = (32 * K) / 8;                           // 16B chunks
#if USE_ASYNC_LDS
    for (int i = tid; i < chunks; i += 128) {
        u32 lds_off = (u32)(i * 16);                     // sB is the only LDS object -> offset 0
        u32 goff    = (u32)(i * 16);
        asm volatile("global_load_async_to_lds_b128 %0, %1, %2"
                     :: "v"(lds_off), "v"(goff), "s"((unsigned long long)gB)
                     : "memory");
    }
    asm volatile("s_wait_asynccnt 0" ::: "memory");
    __syncthreads();
#else
    for (int i = tid; i < chunks; i += 128) {
        uint4 v = *(const uint4*)(gB + i * 8);
        *(uint4*)(sB + i * 8) = v;
    }
    __syncthreads();
#endif

    if (m0 < M) {
        int r = lane & 15, h = lane >> 4;
        const u16* Ap  = A + (size_t)(m0 + r) * K;
        const u16* Bp0 = sB + (size_t)r * K;             // LDS reads (ds ops)
        const u16* Bp1 = sB + (size_t)(16 + r) * K;
        __builtin_prefetch(Ap, 0, 3);
        v8f acc0 = {}, acc1 = {};
        for (int k0 = 0; k0 < K; k0 += 32) {
            v16bf a  = load_frag16x32(Ap,  k0, h);
            v16bf b0 = load_frag16x32(Bp0, k0, h);
            v16bf b1 = load_frag16x32(Bp1, k0, h);
            acc0 = __builtin_amdgcn_wmma_f32_16x16x32_bf16(false, a, false, b0,
                                                           (short)0, acc0, false, false);
            acc1 = __builtin_amdgcn_wmma_f32_16x16x32_bf16(false, a, false, b1,
                                                           (short)0, acc1, false, false);
        }
#pragma unroll
        for (int j = 0; j < 8; ++j) {
            int row = m0 + j + 8 * h;                    // C layout: VGPR j -> M = j + 8h
            int c0 = n0 + r, c1 = n0 + 16 + r;
            float v0 = acc0[j] + (bias ? bias[c0] : 0.f);
            float v1 = acc1[j] + (bias ? bias[c1] : 0.f);
            if (act == 1)      { v0 = fmaxf(v0, 0.f); v1 = fmaxf(v1, 0.f); }
            else if (act == 2) { v0 = tanhf(v0);      v1 = tanhf(v1); }
            C[(size_t)row * N + c0] = v0;
            C[(size_t)row * N + c1] = v1;
        }
    }
}

// ---------------- per-row GEMV: out[i] = x[i,:] . att[:128] ----------------
__global__ void k_gemv128(float* out, const float* x, const float* att, int n) {
    int row = blockIdx.x * 8 + (threadIdx.x >> 5);
    int lane = threadIdx.x & 31;
    if (row >= n) return;
    const float* p = x + (size_t)row * 128;
    float s = 0.f;
    for (int d = lane; d < 128; d += 32) s += p[d] * att[d];
    for (int o = 16; o; o >>= 1) s += __shfl_xor(s, o, 32);
    if (lane == 0) out[row] = s;
}

// ---------------- edge / segment kernels ----------------
__global__ void k_hg_logit(float* lg, const float* px, const float* ph,
                           const int* src, const int* dst, int E) {
    int e = blockIdx.x * blockDim.x + threadIdx.x;
    if (e < E) lg[e] = leaky02(px[src[e]] + ph[dst[e]]);
}
// one wave per edge: e = leaky(xl[src]+xr[dst]) . att
__global__ void k_gat_logit(float* lg, const float* xl, const float* xr,
                            const int* src, const int* dst, const float* att, int E) {
    int e = blockIdx.x * 8 + (threadIdx.x >> 5);
    int lane = threadIdx.x & 31;
    if (e >= E) return;
    const float* pl = xl + (size_t)src[e] * 128;
    const float* pr = xr + (size_t)dst[e] * 128;
    float s = 0.f;
    for (int d = lane; d < 128; d += 32) s += leaky02(pl[d] + pr[d]) * att[d];
    for (int o = 16; o; o >>= 1) s += __shfl_xor(s, o, 32);
    if (lane == 0) lg[e] = s;
}
__global__ void k_seg_max(const float* lg, const int* seg, float* sm, int E) {
    int e = blockIdx.x * blockDim.x + threadIdx.x;
    if (e < E) atomicMaxF(&sm[seg[e]], lg[e]);
}
__global__ void k_seg_exp(float* lg, const int* seg, const float* sm, float* ss, int E) {
    int e = blockIdx.x * blockDim.x + threadIdx.x;
    if (e >= E) return;
    float v = expf(lg[e] - sm[seg[e]]);
    lg[e] = v;
    atomicAdd(&ss[seg[e]], v);
}
__global__ void k_seg_norm(float* lg, const int* seg, const float* ss, int E) {
    int e = blockIdx.x * blockDim.x + threadIdx.x;
    if (e < E) lg[e] = lg[e] / (ss[seg[e]] + 1e-16f);
}
__global__ void k_count(const int* seg, float* cnt, int E) {
    int e = blockIdx.x * blockDim.x + threadIdx.x;
    if (e < E) atomicAdd(&cnt[seg[e]], 1.f);
}
// out[oI[e],:] += (a[e]/div[dI[e]]) * x[iI[e],:]     (2 edges per 256-thread block)
__global__ void k_scatter_hg(float* out, const int* oI, const int* iI,
                             const float* a, const float* x,
                             const float* div, const int* dI, int E) {
    int e = blockIdx.x * 2 + (threadIdx.x >> 7);
    int d = threadIdx.x & 127;
    if (e >= E) return;
    float c = div[dI[e]];
    float w = (c > 0.f) ? a[e] / c : 0.f;
    atomicAdd(&out[(size_t)oI[e] * 128 + d], w * x[(size_t)iI[e] * 128 + d]);
}
// out[dst[e],:] += a[e] * xl[src[e],:]
__global__ void k_scatter_gat(float* out, const int* dst, const int* src,
                              const float* a, const float* xl, int E) {
    int e = blockIdx.x * 2 + (threadIdx.x >> 7);
    int d = threadIdx.x & 127;
    if (e >= E) return;
    atomicAdd(&out[(size_t)dst[e] * 128 + d], a[e] * xl[(size_t)src[e] * 128 + d]);
}
__global__ void k_adj_bits(u32* bits, const int* row, const int* col, int E) {
    int e = blockIdx.x * blockDim.x + threadIdx.x;
    if (e >= E) return;
    int rr = row[e], cc = col[e];
    atomicOr(&bits[(size_t)rr * 32 + (cc >> 5)], 1u << (cc & 31));
}

// ---------------- batchnorm (two-phase column stats) / layernorm ----------------
__global__ void k_colsum_part(const float* x, float* sums, int n) {
    int c  = threadIdx.x;                    // block = 128
    int r0 = blockIdx.x * 256;
    int r1 = min(r0 + 256, n);
    float s = 0.f, s2 = 0.f;
    for (int r = r0; r < r1; ++r) { float v = x[(size_t)r * 128 + c]; s += v; s2 += v * v; }
    atomicAdd(&sums[c], s);
    atomicAdd(&sums[128 + c], s2);
}
__global__ void k_colstats_fin(const float* sums, float* mu, float* var, int n) {
    int c = threadIdx.x;                     // block = 128
    float m = sums[c] / n;
    mu[c] = m;
    var[c] = sums[128 + c] / n - m * m;
}
__global__ void k_bn_apply(float* x, const float* mu, const float* var,
                           const float* g, const float* b, size_t n, int sig) {
    size_t i = (size_t)blockIdx.x * blockDim.x + threadIdx.x;
    if (i >= n) return;
    int c = (int)(i & 127);
    float v = (x[i] - mu[c]) * rsqrtf(var[c] + 1e-5f) * g[c] + b[c];
    if (sig) v = 1.f / (1.f + expf(-v));
    x[i] = v;
}
__global__ void k_ln(float* x, const float* g, const float* b, int n) {
    int row = blockIdx.x * 8 + (threadIdx.x >> 5);
    int lane = threadIdx.x & 31;
    if (row >= n) return;
    float* p = x + (size_t)row * 128;
    float s = 0.f, s2 = 0.f;
    for (int d = lane; d < 128; d += 32) { float v = p[d]; s += v; s2 += v * v; }
    for (int o = 16; o; o >>= 1) { s += __shfl_xor(s, o, 32); s2 += __shfl_xor(s2, o, 32); }
    float m = s / 128.f;
    float inv = rsqrtf(s2 / 128.f - m * m + 1e-5f);
    for (int d = lane; d < 128; d += 32) p[d] = (p[d] - m) * inv * g[d] + b[d];
}

// ---------------- fused masked attention ----------------
// out[nq,128] = where(anymask, softmax_j(mask ? q.k/sqrtD : -1e9) @ kemb, 0)
// block = 256 (8 waves) handles 16 query rows; scores staged in LDS as bf16 (32KB).
__global__ void k_masked_attn(float* __restrict__ out, const u16* __restrict__ qb,
                              const u16* __restrict__ kemb_b,   // [1024 x 128] bf16
                              const u16* __restrict__ kembT_b,  // [128 x 1024] bf16
                              const u32* __restrict__ adjbits,  // [nq x 32]
                              int nq) {
    __shared__ u16   sc[16 * 1024];
    __shared__ float red[256];
    __shared__ float rstat[16];
    int wave = threadIdx.x >> 5, lane = threadIdx.x & 31;
    int q0 = blockIdx.x * 16;
    int r = lane & 15, h = lane >> 4;
    const float scale = 0.08838834764831845f;   // 1/sqrt(128)

    // phase 1: S = Q K^T / sqrt(D). Key-tiles jt and jt+8 paired per iteration:
    // shared Q fragment + two independent accumulator chains.
    const u16* Ap = qb + (size_t)(q0 + r) * 128;
    for (int jt = wave; jt < 64; jt += 16) {
        const u16* Bp0 = kemb_b + (size_t)(jt * 16 + r) * 128;
        const u16* Bp1 = kemb_b + (size_t)((jt + 8) * 16 + r) * 128;
        v8f acc0 = {}, acc1 = {};
        for (int k0 = 0; k0 < 128; k0 += 32) {
            v16bf a  = load_frag16x32(Ap,  k0, h);
            v16bf b0 = load_frag16x32(Bp0, k0, h);
            v16bf b1 = load_frag16x32(Bp1, k0, h);
            acc0 = __builtin_amdgcn_wmma_f32_16x16x32_bf16(false, a, false, b0,
                                                           (short)0, acc0, false, false);
            acc1 = __builtin_amdgcn_wmma_f32_16x16x32_bf16(false, a, false, b1,
                                                           (short)0, acc1, false, false);
        }
#pragma unroll
        for (int j = 0; j < 8; ++j) {
            sc[(j + 8 * h) * 1024 + jt * 16 + r]       = f2bf(acc0[j] * scale);
            sc[(j + 8 * h) * 1024 + (jt + 8) * 16 + r] = f2bf(acc1[j] * scale);
        }
    }
    __syncthreads();

    // phase 2: masked softmax. 16 threads per row, 64 keys (2 mask words) each.
    {
        int row = threadIdx.x >> 4;              // 0..15
        int sub = threadIdx.x & 15;              // 0..15
        const u32* bits = adjbits + (size_t)(q0 + row) * 32;
        u32 b0 = bits[sub * 2], b1 = bits[sub * 2 + 1];
        u16* srow = sc + row * 1024;
        int jb = sub * 64;

        float m = -1e30f;
        for (int bpos = 0; bpos < 32; ++bpos) {
            if ((b0 >> bpos) & 1) m = fmaxf(m, bf2f(srow[jb + bpos]));
            if ((b1 >> bpos) & 1) m = fmaxf(m, bf2f(srow[jb + 32 + bpos]));
        }
        red[threadIdx.x] = m;
        __syncthreads();
        if (sub == 0) {
            float mm = -1e30f;
            for (int i = 0; i < 16; ++i) mm = fmaxf(mm, red[row * 16 + i]);
            rstat[row] = mm;
        }
        __syncthreads();
        float mm = rstat[row];

        float ssum = 0.f;
        for (int bpos = 0; bpos < 32; ++bpos) {
            int j = jb + bpos; float p = 0.f;
            if ((b0 >> bpos) & 1) { p = expf(bf2f(srow[j]) - mm); ssum += p; }
            srow[j] = f2bf(p);
            j = jb + 32 + bpos; p = 0.f;
            if ((b1 >> bpos) & 1) { p = expf(bf2f(srow[j]) - mm); ssum += p; }
            srow[j] = f2bf(p);
        }
        red[threadIdx.x] = ssum;
        __syncthreads();
        if (sub == 0) {
            float tot = 0.f;
            for (int i = 0; i < 16; ++i) tot += red[row * 16 + i];
            rstat[row] = (tot > 0.f) ? 1.f / tot : 0.f;   // empty row -> all-zero output
        }
        __syncthreads();
        float inv = rstat[row];
        for (int j = jb; j < jb + 64; ++j) srow[j] = f2bf(bf2f(srow[j]) * inv);
    }
    __syncthreads();

    // phase 3: out = P @ kemb  (wave w -> output cols [16w,16w+16)).
    // K=1024 reduction split into two interleaved accumulator chains.
    int d0 = wave * 16;
    const u16* Bp2 = kembT_b + (size_t)(d0 + r) * 1024;
    const u16* Pp  = sc + r * 1024;
    v8f acc0 = {}, acc1 = {};
    for (int k0 = 0; k0 < 1024; k0 += 64) {
        v16bf a0 = load_frag16x32(Pp,  k0,      h);   // LDS reads (ds ops)
        v16bf b0 = load_frag16x32(Bp2, k0,      h);
        v16bf a1 = load_frag16x32(Pp,  k0 + 32, h);
        v16bf b1 = load_frag16x32(Bp2, k0 + 32, h);
        acc0 = __builtin_amdgcn_wmma_f32_16x16x32_bf16(false, a0, false, b0,
                                                       (short)0, acc0, false, false);
        acc1 = __builtin_amdgcn_wmma_f32_16x16x32_bf16(false, a1, false, b1,
                                                       (short)0, acc1, false, false);
    }
#pragma unroll
    for (int j = 0; j < 8; ++j)
        out[(size_t)(q0 + j + 8 * h) * 128 + d0 + r] = acc0[j] + acc1[j];
}

// ---------------- fused SSL: sc = A B^T, reduce softplus ----------------
// 8 waves per block; each wave computes TWO adjacent 16x16 score tiles
// (shared A fragment, two accumulator chains) and reduces them in-register.
// accum[0] += sum_diag softplus(-sc_ii) ; accum[1] += sum_offdiag softplus(sc_ij) + N*ln2
__global__ void k_ssl(const u16* __restrict__ Ab, const u16* __restrict__ Bb,
                      float* accum, int N) {
    int wave = threadIdx.x >> 5, lane = threadIdx.x & 31;
    int m0 = (blockIdx.y * 8 + wave) * 16;
    int n0 = blockIdx.x * 32;
    if (m0 >= N) return;
    int r = lane & 15, h = lane >> 4;
    const u16* Ap  = Ab + (size_t)(m0 + r) * 128;
    const u16* Bp0 = Bb + (size_t)(n0 + r) * 128;
    const u16* Bp1 = Bb + (size_t)(n0 + 16 + r) * 128;
    v8f acc0 = {}, acc1 = {};
    for (int k0 = 0; k0 < 128; k0 += 32) {
        v16bf a  = load_frag16x32(Ap,  k0, h);
        v16bf b0 = load_frag16x32(Bp0, k0, h);
        v16bf b1 = load_frag16x32(Bp1, k0, h);
        acc0 = __builtin_amdgcn_wmma_f32_16x16x32_bf16(false, a, false, b0,
                                                       (short)0, acc0, false, false);
        acc1 = __builtin_amdgcn_wmma_f32_16x16x32_bf16(false, a, false, b1,
                                                       (short)0, acc1, false, false);
    }
    float lp = 0.f, ln_ = 0.f;
#pragma unroll
    for (int j = 0; j < 8; ++j) {
        int row = m0 + j + 8 * h;
        int c0 = n0 + r, c1 = n0 + 16 + r;
        float v0 = acc0[j], v1 = acc1[j];
        if (row == c0) { lp += softplusf(-v0); ln_ += 0.6931471805599453f; }
        else           { ln_ += softplusf(v0); }
        if (row == c1) { lp += softplusf(-v1); ln_ += 0.6931471805599453f; }
        else           { ln_ += softplusf(v1); }
    }
    for (int o = 16; o; o >>= 1) { lp += __shfl_xor(lp, o, 32); ln_ += __shfl_xor(ln_, o, 32); }
    if (lane == 0) { atomicAdd(&accum[0], lp); atomicAdd(&accum[1], ln_); }
}
__global__ void k_fincon(float* dst, const float* acc, float invN, float invN2) {
    if (threadIdx.x == 0) dst[0] = acc[0] * invN + acc[1] * invN2;
}

// ---------------- final batch gather ----------------
__global__ void k_gather_batch(float* out, const float* uo, const float* eo,
                               const int* sid, const int* eid) {
    int b = blockIdx.x, d = threadIdx.x;    // block = 256
    float v = (d < 128) ? uo[(size_t)sid[b] * 128 + d]
                        : eo[(size_t)eid[b] * 128 + (d - 128)];
    out[(size_t)b * 256 + d] = v;
}

// ================= host-side orchestration =================
struct Alloc {
    char* base; size_t off;
    void* take(size_t bytes) {
        size_t o = (off + 255) & ~size_t(255);
        off = o + bytes;
        return base + o;
    }
};

static inline void fillf(hipStream_t s, float* p, float v, size_t n) {
    k_fill_f32<<<dim3((unsigned)((n + 255) / 256)), dim3(256), 0, s>>>(p, v, n);
}
static inline void tobf(hipStream_t s, u16* d, const float* x, size_t n) {
    k_to_bf16<<<dim3((unsigned)((n + 255) / 256)), dim3(256), 0, s>>>(d, x, n);
}
static inline void gemm(hipStream_t s, float* C, const u16* A, const u16* Bt,
                        const float* bias, int M, int N, int K, int act) {
    dim3 g(N / 32, (M / 16 + 3) / 4);
    k_gemm_bf16<<<g, dim3(128), 0, s>>>(C, A, Bt, bias, M, N, K, act);
}
static inline void seg_softmax(hipStream_t s, Alloc& a, float* lg, const int* seg,
                               int nseg, int E) {
    float* sm = (float*)a.take((size_t)nseg * 4);
    float* ss = (float*)a.take((size_t)nseg * 4);
    fillf(s, sm, -1e30f, nseg); fillf(s, ss, 0.f, nseg);
    dim3 ge((E + 255) / 256), be(256);
    k_seg_max <<<ge, be, 0, s>>>(lg, seg, sm, E);
    k_seg_exp <<<ge, be, 0, s>>>(lg, seg, sm, ss, E);
    k_seg_norm<<<ge, be, 0, s>>>(lg, seg, ss, E);
}
static inline void colstats(hipStream_t s, Alloc& a, const float* x,
                            float* mu, float* var, int n) {
    float* sums = (float*)a.take(256 * 4);
    fillf(s, sums, 0.f, 256);
    k_colsum_part<<<dim3((n + 255) / 256), dim3(128), 0, s>>>(x, sums, n);
    k_colstats_fin<<<dim3(1), dim3(128), 0, s>>>(sums, mu, var, n);
}

// one hypergraph conv layer
static void run_hg(hipStream_t s, Alloc a, const float* x, const float* he,
                   const int* src, const int* dst, int n, int m, int E,
                   const u16* Wt, const float* att, const float* bias,
                   float* out, int relu) {
    u16*  xb = (u16*) a.take((size_t)n * 128 * 2);
    u16*  hb = (u16*) a.take((size_t)m * 128 * 2);
    float* xp = (float*)a.take((size_t)n * 128 * 4);
    float* hp = (float*)a.take((size_t)m * 128 * 4);
    float* px = (float*)a.take((size_t)n * 4);
    float* ph = (float*)a.take((size_t)m * 4);
    float* lg = (float*)a.take((size_t)E * 4);
    float* dn = (float*)a.take((size_t)n * 4);
    float* bm = (float*)a.take((size_t)m * 4);
    float* ef = (float*)a.take((size_t)m * 128 * 4);

    tobf(s, xb, x, (size_t)n * 128); tobf(s, hb, he, (size_t)m * 128);
    gemm(s, xp, xb, Wt, nullptr, n, 128, 128, 0);
    gemm(s, hp, hb, Wt, nullptr, m, 128, 128, 0);
    k_gemv128<<<dim3((n + 7) / 8), dim3(256), 0, s>>>(px, xp, att, n);
    k_gemv128<<<dim3((m + 7) / 8), dim3(256), 0, s>>>(ph, hp, att + 128, m);
    dim3 ge((E + 255) / 256), be(256);
    k_hg_logit<<<ge, be, 0, s>>>(lg, px, ph, src, dst, E);
    seg_softmax(s, a, lg, src, n, E);
    fillf(s, dn, 0.f, n); fillf(s, bm, 0.f, m);
    k_count<<<ge, be, 0, s>>>(src, dn, E);
    k_count<<<ge, be, 0, s>>>(dst, bm, E);
    fillf(s, ef, 0.f, (size_t)m * 128);
    k_scatter_hg<<<dim3((E + 1) / 2), dim3(256), 0, s>>>(ef, dst, src, lg, xp, bm, dst, E);
    k_rows_bias<<<dim3(n), dim3(128), 0, s>>>(out, bias, n);
    k_scatter_hg<<<dim3((E + 1) / 2), dim3(256), 0, s>>>(out, src, dst, lg, ef, dn, src, E);
    if (relu) k_relu<<<dim3((unsigned)(((size_t)n * 128 + 255) / 256)), dim3(256), 0, s>>>(out, (size_t)n * 128);
}

static void run_gat_layer(hipStream_t s, Alloc a, const float* x,
                          const int* src, const int* dst, int n, int E,
                          const u16* WlT, const u16* WrT, const float* att,
                          const float* bias, float* out, int relu) {
    u16*  xb = (u16*) a.take((size_t)n * 128 * 2);
    float* xl = (float*)a.take((size_t)n * 128 * 4);
    float* xr = (float*)a.take((size_t)n * 128 * 4);
    float* lg = (float*)a.take((size_t)E * 4);
    tobf(s, xb, x, (size_t)n * 128);
    gemm(s, xl, xb, WlT, nullptr, n, 128, 128, 0);
    gemm(s, xr, xb, WrT, nullptr, n, 128, 128, 0);
    k_gat_logit<<<dim3((E + 7) / 8), dim3(256), 0, s>>>(lg, xl, xr, src, dst, att, E);
    seg_softmax(s, a, lg, dst, n, E);
    k_rows_bias<<<dim3(n), dim3(128), 0, s>>>(out, bias, n);
    k_scatter_gat<<<dim3((E + 1) / 2), dim3(256), 0, s>>>(out, dst, src, lg, xl, E);
    if (relu) k_relu<<<dim3((unsigned)(((size_t)n * 128 + 255) / 256)), dim3(256), 0, s>>>(out, (size_t)n * 128);
}

static void run_gat_stack(hipStream_t s, Alloc a, const float* x, float* out,
                          const int* src, const int* dst, int n, int E,
                          const u16* glWt, const u16* grWt,
                          const float* att, const float* bias) {
    float* tmp = (float*)a.take((size_t)n * 128 * 4);
    run_gat_layer(s, a, x,   src, dst, n, E, glWt,             grWt,             att,       bias,       tmp, 1);
    run_gat_layer(s, a, tmp, src, dst, n, E, glWt + 128 * 128, grWt + 128 * 128, att + 128, bias + 128, out, 0);
}

extern "C" void kernel_launch(void* const* d_in, const int* in_sizes, int n_in,
                              void* d_out, int out_size, void* d_ws, size_t ws_size,
                              hipStream_t stream) {
    (void)in_sizes; (void)n_in; (void)ws_size;
    hipStream_t s = stream;

    const int*   stu_id   = (const int*)  d_in[0];
    const int*   exer_id  = (const int*)  d_in[1];
    const int*   ei_sk    = (const int*)  d_in[2];
    const int*   ei_ek    = (const int*)  d_in[3];
    const int*   ei_kkd   = (const int*)  d_in[4];
    const int*   ei_kku   = (const int*)  d_in[5];
    const int*   ei_ee    = (const int*)  d_in[6];
    const int*   ei_ss    = (const int*)  d_in[7];
    const float* emb_stu  = (const float*)d_in[8];
    const float* emb_exer = (const float*)d_in[9];
    const float* emb_kn   = (const float*)d_in[10];
    const float* hg_W     = (const float*)d_in[11];
    const float* hg_att   = (const float*)d_in[12];
    const float* hg_b     = (const float*)d_in[13];
    const float* gat_Wl   = (const float*)d_in[14];
    const float* gat_Wr   = (const float*)d_in[15];
    const float* gat_att  = (const float*)d_in[16];
    const float* gat_b    = (const float*)d_in[17];
    const float* fcs_W    = (const float*)d_in[18];
    const float* fcs_b    = (const float*)d_in[19];
    const float* fc6_W    = (const float*)d_in[20];
    const float* fc6_b    = (const float*)d_in[21];
    const float* bn_g     = (const float*)d_in[22];
    const float* bn_b     = (const float*)d_in[23];
    const float* ln_g     = (const float*)d_in[24];
    const float* ln_b     = (const float*)d_in[25];

    const int *sk_s = ei_sk, *sk_d = ei_sk + ESK;
    const int *ek_s = ei_ek, *ek_d = ei_ek + EEK;

    Alloc A{(char*)d_ws, 0};

    // persistent bf16 weights (pre-transposed B^T operands)
    u16* hgWt  = (u16*)A.take((size_t)8 * 128 * 128 * 2);
    u16* glWt  = (u16*)A.take((size_t)8 * 128 * 128 * 2);
    u16* grWt  = (u16*)A.take((size_t)8 * 128 * 128 * 2);
    u16* fcsWb = (u16*)A.take((size_t)5 * 128 * 256 * 2);
    u16* fc6Wb = (u16*)A.take((size_t)128 * 128 * 2);
    // persistent f32 activations
    float* s_hg    = (float*)A.take((size_t)Sn  * 128 * 4);
    float* e_hg    = (float*)A.take((size_t)ENn * 128 * 4);
    float* k_s     = (float*)A.take((size_t)Kn  * 128 * 4);
    float* k_e     = (float*)A.take((size_t)Kn  * 128 * 4);
    float* k_dir   = (float*)A.take((size_t)Kn  * 128 * 4);
    float* k_und   = (float*)A.take((size_t)Kn  * 128 * 4);
    float* k1      = (float*)A.take((size_t)Kn  * 128 * 4);
    float* k2      = (float*)A.take((size_t)Kn  * 128 * 4);
    float* k_sg    = (float*)A.take((size_t)Kn  * 128 * 4);
    float* e_sg    = (float*)A.take((size_t)ENn * 128 * 4);
    float* s_sg    = (float*)A.take((size_t)Sn  * 128 * 4);
    float* exer_sg = (float*)A.take((size_t)ENn * 128 * 4);
    float* stu_sg  = (float*)A.take((size_t)Sn  * 128 * 4);
    float* exer_o  = (float*)A.take((size_t)ENn * 128 * 4);
    float* user_o  = (float*)A.take((size_t)Sn  * 128 * 4);
    u16* ksg_b  = (u16*)A.take((size_t)Kn * 128 * 2);
    u16* ksgT_b = (u16*)A.take((size_t)128 * Kn * 2);
    u32* bits_ek = (u32*)A.take((size_t)ENn * 32 * 4);
    u32* bits_sk = (u32*)A.take((size_t)Sn  * 32 * 4);
    float* accum = (float*)A.take(2 * 4);
    u16* exsg_b = (u16*)A.take((size_t)ENn * 128 * 2);
    u16* ehg_b  = (u16*)A.take((size_t)ENn * 128 * 2);
    u16* stsg_b = (u16*)A.take((size_t)Sn  * 128 * 2);
    u16* shg_b  = (u16*)A.take((size_t)Sn  * 128 * 2);

    // ---- weight prep: transpose/convert to bf16 ----
    dim3 wt((128 * 128 + 255) / 256), wb(256);
    for (int i = 0; i < 8; ++i) {
        k_transpose_bf16<<<wt, wb, 0, s>>>(hgWt + (size_t)i * 16384, hg_W   + (size_t)i * 16384, 128, 128);
        k_transpose_bf16<<<wt, wb, 0, s>>>(glWt + (size_t)i * 16384, gat_Wl + (size_t)i * 16384, 128, 128);
        k_transpose_bf16<<<wt, wb, 0, s>>>(grWt + (size_t)i * 16384, gat_Wr + (size_t)i * 16384, 128, 128);
    }
    tobf(s, fcsWb, fcs_W, (size_t)5 * 128 * 256);   // _lin uses W.T => Bt = W as stored
    tobf(s, fc6Wb, fc6_W, (size_t)128 * 128);

    // ---- hypergraph convolutions ----
    { Alloc a = A; float* tmp = (float*)a.take((size_t)Sn * 128 * 4);
      run_hg(s, a, emb_stu, emb_kn, sk_s, sk_d, Sn, Kn, ESK, hgWt,             hg_att,       hg_b,       tmp,  0);
      run_hg(s, a, tmp,     emb_kn, sk_s, sk_d, Sn, Kn, ESK, hgWt + 16384,     hg_att + 256, hg_b + 128, s_hg, 1); }
    { Alloc a = A; float* tmp = (float*)a.take((size_t)Kn * 128 * 4);
      run_hg(s, a, emb_kn, s_hg, sk_d, sk_s, Kn, Sn, ESK, hgWt + 2 * 16384, hg_att + 2 * 256, hg_b + 2 * 128, tmp, 0);
      run_hg(s, a, tmp,    s_hg, sk_d, sk_s, Kn, Sn, ESK, hgWt + 3 * 16384, hg_att + 3 * 256, hg_b + 3 * 128, k_s, 1); }
    { Alloc a = A; float* tmp = (float*)a.take((size_t)ENn * 128 * 4);
      run_hg(s, a, emb_exer, emb_kn, ek_s, ek_d, ENn, Kn, EEK, hgWt + 4 * 16384, hg_att + 4 * 256, hg_b + 4 * 128, tmp,  0);
      run_hg(s, a, tmp,      emb_kn, ek_s, ek_d, ENn, Kn, EEK, hgWt + 5 * 16384, hg_att + 5 * 256, hg_b + 5 * 128, e_hg, 1); }
    { Alloc a = A; float* tmp = (float*)a.take((size_t)Kn * 128 * 4);
      run_hg(s, a, emb_kn, e_hg, ek_d, ek_s, Kn, ENn, EEK, hgWt + 6 * 16384, hg_att + 6 * 256, hg_b + 6 * 128, tmp, 0);
      run_hg(s, a, tmp,    e_hg, ek_d, ek_s, Kn, ENn, EEK, hgWt + 7 * 16384, hg_att + 7 * 256, hg_b + 7 * 128, k_e, 1); }

    // ---- GAT stacks ----
    { Alloc a = A; run_gat_stack(s, a, emb_kn, k_dir, ei_kkd, ei_kkd + EKK, Kn, EKK,
                                 glWt + 0 * 2 * 16384, grWt + 0 * 2 * 16384, gat_att + 0 * 256, gat_b + 0 * 256); }
    { Alloc a = A; run_gat_stack(s, a, emb_kn, k_und, ei_kku, ei_kku + EKK, Kn, EKK,
                                 glWt + 1 * 2 * 16384, grWt + 1 * 2 * 16384, gat_att + 1 * 256, gat_b + 1 * 256); }
    { Alloc a = A; run_gat_stack(s, a, emb_exer, e_sg, ei_ee, ei_ee + EEE, ENn, EEE,
                                 glWt + 2 * 2 * 16384, grWt + 2 * 2 * 16384, gat_att + 2 * 256, gat_b + 2 * 256); }
    { Alloc a = A; run_gat_stack(s, a, emb_stu, s_sg, ei_ss, ei_ss + ESS, Sn, ESS,
                                 glWt + 3 * 2 * 16384, grWt + 3 * 2 * 16384, gat_att + 3 * 256, gat_b + 3 * 256); }

    // ---- knowledge fusion: k1 / k2 / k_sg ----
    { Alloc a = A;
      u16* cc = (u16*)a.take((size_t)Kn * 256 * 2);
      float* mu = (float*)a.take(128 * 4); float* var = (float*)a.take(128 * 4);
      k_concat2_bf16<<<dim3((Kn * 256 + 255) / 256), dim3(256), 0, s>>>(cc, k_s, k_e, Kn);
      gemm(s, k1, cc, fcsWb + (size_t)2 * 128 * 256, fcs_b + 2 * 128, Kn, 128, 256, 0);
      colstats(s, a, k1, mu, var, Kn);
      k_bn_apply<<<dim3((Kn * 128 + 255) / 256), dim3(256), 0, s>>>(k1, mu, var, bn_g, bn_b, (size_t)Kn * 128, 0);
      k_concat2_bf16<<<dim3((Kn * 256 + 255) / 256), dim3(256), 0, s>>>(cc, k_dir, k_und, Kn);
      gemm(s, k2, cc, fcsWb + (size_t)3 * 128 * 256, fcs_b + 3 * 128, Kn, 128, 256, 0);
      colstats(s, a, k2, mu, var, Kn);
      k_bn_apply<<<dim3((Kn * 128 + 255) / 256), dim3(256), 0, s>>>(k2, mu, var, bn_g + 128, bn_b + 128, (size_t)Kn * 128, 0);
      k_concat2_bf16<<<dim3((Kn * 256 + 255) / 256), dim3(256), 0, s>>>(cc, k1, k2, Kn);
      gemm(s, k_sg, cc, fcsWb + (size_t)4 * 128 * 256, fcs_b + 4 * 128, Kn, 128, 256, 0);
      k_ln<<<dim3((Kn + 7) / 8), dim3(256), 0, s>>>(k_sg, ln_g, ln_b, Kn);
    }
    tobf(s, ksg_b, k_sg, (size_t)Kn * 128);
    k_transpose_bf16<<<dim3((Kn * 128 + 255) / 256), dim3(256), 0, s>>>(ksgT_b, k_sg, Kn, 128);

    // ---- adjacency bitmasks ----
    k_fill_u32<<<dim3((ENn * 32 + 255) / 256), dim3(256), 0, s>>>(bits_ek, 0u, (size_t)ENn * 32);
    k_fill_u32<<<dim3((Sn  * 32 + 255) / 256), dim3(256), 0, s>>>(bits_sk, 0u, (size_t)Sn * 32);
    k_adj_bits<<<dim3((EEK + 255) / 256), dim3(256), 0, s>>>(bits_ek, ek_s, ek_d, EEK);
    k_adj_bits<<<dim3((ESK + 255) / 256), dim3(256), 0, s>>>(bits_sk, sk_s, sk_d, ESK);

    // ---- exercise branch: masked attention + fc6 + bn + sigmoid ----
    { Alloc a = A;
      u16*  qb  = (u16*) a.take((size_t)ENn * 128 * 2);
      float* at = (float*)a.take((size_t)ENn * 128 * 4);
      u16*  ab  = (u16*) a.take((size_t)ENn * 128 * 2);
      float* mu = (float*)a.take(128 * 4); float* var = (float*)a.take(128 * 4);
      tobf(s, qb, e_sg, (size_t)ENn * 128);
      k_masked_attn<<<dim3(ENn / 16), dim3(256), 0, s>>>(at, qb, ksg_b, ksgT_b, bits_ek, ENn);
      tobf(s, ab, at, (size_t)ENn * 128);
      gemm(s, exer_sg, ab, fc6Wb, fc6_b, ENn, 128, 128, 0);
      colstats(s, a, exer_sg, mu, var, ENn);
      k_bn_apply<<<dim3((unsigned)(((size_t)ENn * 128 + 255) / 256)), dim3(256), 0, s>>>(
          exer_sg, mu, var, bn_g + 256, bn_b + 256, (size_t)ENn * 128, 1);
    }
    // ---- student branch: masked attention + sigmoid ----
    { Alloc a = A;
      u16* qb = (u16*)a.take((size_t)Sn * 128 * 2);
      tobf(s, qb, s_sg, (size_t)Sn * 128);
      k_masked_attn<<<dim3(Sn / 16), dim3(256), 0, s>>>(stu_sg, qb, ksg_b, ksgT_b, bits_sk, Sn);
      k_sigmoid<<<dim3((unsigned)(((size_t)Sn * 128 + 255) / 256)), dim3(256), 0, s>>>(stu_sg, (size_t)Sn * 128);
    }

    // ---- fused SSL contrastive loss ----
    tobf(s, exsg_b, exer_sg, (size_t)ENn * 128);
    tobf(s, ehg_b,  e_hg,    (size_t)ENn * 128);
    tobf(s, stsg_b, stu_sg,  (size_t)Sn  * 128);
    tobf(s, shg_b,  s_hg,    (size_t)Sn  * 128);
    fillf(s, accum, 0.f, 2);
    k_ssl<<<dim3(ENn / 32, (ENn / 16 + 7) / 8), dim3(256), 0, s>>>(exsg_b, ehg_b, accum, ENn);
    k_ssl<<<dim3(Sn  / 32, (Sn  / 16 + 7) / 8), dim3(256), 0, s>>>(stsg_b, shg_b, accum, Sn);

    // ---- output heads ----
    { Alloc a = A;
      u16* cc = (u16*)a.take((size_t)ENn * 256 * 2);
      k_concat2_bf16<<<dim3((unsigned)(((size_t)ENn * 256 + 255) / 256)), dim3(256), 0, s>>>(cc, exer_sg, e_hg, ENn);
      gemm(s, exer_o, cc, fcsWb + (size_t)0 * 128 * 256, fcs_b + 0 * 128, ENn, 128, 256, 2);
      k_concat2_bf16<<<dim3((unsigned)(((size_t)Sn * 256 + 255) / 256)), dim3(256), 0, s>>>(cc, stu_sg, s_hg, Sn);
      gemm(s, user_o, cc, fcsWb + (size_t)1 * 128 * 256, fcs_b + 1 * 128, Sn, 128, 256, 2);
    }

    float* out = (float*)d_out;
    k_gather_batch<<<dim3(Bn), dim3(256), 0, s>>>(out, user_o, exer_o, stu_id, exer_id);
    k_fincon<<<dim3(1), dim3(1), 0, s>>>(out + (out_size - 1), accum,
                                         1.f / 8000.f, 1.f / (8000.f * 8000.f));
}
